// Model_79860621902384
// MI455X (gfx1250) — compile-verified, matrix operand order checked by reading, not run on hardware
//
#include <hip/hip_runtime.h>
#include <hip/hip_bf16.h>

// ---------------------------------------------------------------------------
// DeepGate-style levelized GNN on MI455X (gfx1250), bf16 WMMA + f32 accumulate
// ---------------------------------------------------------------------------

#define DHID 128
#define DXF   64
#define NLVL  12
#define NNODE 80000
#define NEDGE 200000

typedef __attribute__((ext_vector_type(16))) __bf16 v16bf;
typedef __attribute__((ext_vector_type(8)))  __bf16 v8bf;
typedef __attribute__((ext_vector_type(8)))  float  v8f;

// counter block layout (ints): [0..12]=offN, [16..28]=offE, [32..43]=curN,
// [48..59]=curE, [64..75]=cntN, [80..91]=cntE
#define C_OFFN 0
#define C_OFFE 16
#define C_CURN 32
#define C_CURE 48
#define C_CNTN 64
#define C_CNTE 80

// ---- WMMA fragment loader -------------------------------------------------
// A (16x32 MxK, 16-bit): lane m(0..15) holds row m, K = {kt*32+0..7, kt*32+16..23};
// lanes 16..31 hold row m with K = {kt*32+8..15, kt*32+24..31}.
// B (32x16 KxN) is symmetric with lane = column n; since gemm is x @ W^T,
// B column n over k == row n of W (contiguous), so the same loader serves both.
__device__ __forceinline__ v16bf load_frag(const __bf16* row, int kt, int lane) {
  const int hs8 = (lane >> 4) * 8;
  const int k0  = kt * 32 + hs8;
  const int k1  = k0 + 16;
  v8bf lo = *(const v8bf*)(row + k0);
  v8bf hi = *(const v8bf*)(row + k1);
  v16bf a;
#pragma unroll
  for (int i = 0; i < 8; ++i) { a[i] = lo[i]; a[i + 8] = hi[i]; }
  return a;
}

__device__ __forceinline__ v8f wmma_bf16(v16bf a, v16bf b, v8f c) {
  return __builtin_amdgcn_wmma_f32_16x16x32_bf16(false, a, false, b, (short)0, c,
                                                 false, false);
}

__device__ __forceinline__ float sigmoidf_(float x) {
  return 1.0f / (1.0f + __expf(-x));
}

// ---- partition kernels ----------------------------------------------------
__global__ void k_zero_counts(int* c) {
  if (threadIdx.x < 128) c[threadIdx.x] = 0;
}

__global__ void k_count(const int* fl, const int* esrc, const int* edst, int* c) {
  int i = blockIdx.x * blockDim.x + threadIdx.x;
  if (i < NNODE) atomicAdd(&c[C_CNTN + fl[i]], 1);
  if (i < NEDGE) atomicAdd(&c[C_CNTE + fl[edst[i]]], 1);
}

__global__ void k_offsets(int* c) {
  if (threadIdx.x != 0 || blockIdx.x != 0) return;
  c[C_OFFN] = 0;
  c[C_OFFE] = 0;
  for (int l = 0; l < NLVL; ++l) {
    c[C_OFFN + l + 1] = c[C_OFFN + l] + c[C_CNTN + l];
    c[C_OFFE + l + 1] = c[C_OFFE + l] + c[C_CNTE + l];
    c[C_CURN + l] = c[C_OFFN + l];
    c[C_CURE + l] = c[C_OFFE + l];
  }
}

__global__ void k_fill(const int* fl, const int* esrc, const int* edst, int* c,
                       int* lvlNodes, int* lvlSrc, int* lvlDst) {
  int i = blockIdx.x * blockDim.x + threadIdx.x;
  if (i < NNODE) {
    int l = fl[i];
    int p = atomicAdd(&c[C_CURN + l], 1);
    lvlNodes[p] = i;
  }
  if (i < NEDGE) {
    int l = fl[edst[i]];
    int p = atomicAdd(&c[C_CURE + l], 1);
    lvlSrc[p] = esrc[i];
    lvlDst[p] = edst[i];
  }
}

// ---- init: PI encode + zero state, build bf16 mirrors ---------------------
__global__ void k_init(const float* hs_init, const int* fl, float* hs_f32,
                       float* hf_f32, __bf16* hs_bf, __bf16* hf_bf,
                       __bf16* ns_bf, float* msg_s, float* msg_f) {
  int i = blockIdx.x * blockDim.x + threadIdx.x;
  if (i >= NNODE) return;
  const bool pi = (fl[i] == 0);
  const float* r = hs_init + (size_t)i * DHID;
  float s = 0.f;
  if (pi) {
    for (int k = 0; k < DHID; ++k) s += r[k] * r[k];
  }
  float inv = pi ? rsqrtf(s) : 0.f;
  size_t b = (size_t)i * DHID;
  size_t b2 = (size_t)i * (2 * DHID);
  for (int k = 0; k < DHID; ++k) {
    float v = pi ? r[k] * inv : 0.f;
    hs_f32[b + k] = v;
    hf_f32[b + k] = 0.f;
    hs_bf[b + k] = (__bf16)v;
    hf_bf[b + k] = (__bf16)0.f;
    ns_bf[b2 + k] = (__bf16)v;          // node_state = [hs_round_start | hf=0]
    ns_bf[b2 + DHID + k] = (__bf16)0.f;
    msg_s[b + k] = 0.f;
    msg_f[b + k] = 0.f;
  }
}

__global__ void k_cvt(const float* s, __bf16* d, int n) {
  int i = blockIdx.x * blockDim.x + threadIdx.x;
  if (i < n) d[i] = (__bf16)s[i];
}

// ---- edge MLP3 + segment scatter ------------------------------------------
// 16 edges per block, 8 waves each owning one 16-wide output column tile.
template <int INK>
__global__ __launch_bounds__(256) void k_msg(
    int level, const int* c, const int* lvlSrc, const int* lvlDst,
    const __bf16* in, const __bf16* W0, const float* b0, const __bf16* W1,
    const float* b1, const __bf16* W2, const float* b2, float* msg) {
  __shared__ __bf16 sIn[16 * INK];
  __shared__ __bf16 sT1[16 * DHID];
  __shared__ __bf16 sT2[16 * DHID];
  __shared__ int sSrc[16];
  __shared__ int sDst[16];

  const int off = c[C_OFFE + level];
  const int cnt = c[C_OFFE + level + 1] - off;
  const int t = blockIdx.x;
  if (t * 16 >= cnt) return;

  const int tid = threadIdx.x;
  const int wave = tid >> 5;
  const int lane = tid & 31;

  if (tid < 16) {
    int e = t * 16 + tid;
    bool ok = (e < cnt);
    sSrc[tid] = ok ? lvlSrc[off + e] : -1;
    sDst[tid] = ok ? lvlDst[off + e] : -1;
  }
  __syncthreads();

  // gather source rows (bf16) into LDS, zero-fill invalid rows
  {
    const int chunks = 16 * INK / 8;
#pragma unroll
    for (int p = tid; p < chunks; p += 256) {
      int m = p / (INK / 8);
      int cb = p % (INK / 8);
      int s = sSrc[m];
      v8bf v;
      if (s >= 0) {
        v = *(const v8bf*)(in + (size_t)s * INK + cb * 8);
      } else {
#pragma unroll
        for (int q = 0; q < 8; ++q) v[q] = (__bf16)0.f;
      }
      *(v8bf*)(sIn + m * INK + cb * 8) = v;
    }
  }
  __syncthreads();

  const int n = wave * 16 + (lane & 15);
  const int rbase = (lane < 16) ? 0 : 8;

  // layer 1: sIn[16xINK] * W0^T -> relu -> sT1
  {
    v8f acc = {0.f, 0.f, 0.f, 0.f, 0.f, 0.f, 0.f, 0.f};
    const __bf16* arow = sIn + (size_t)(lane & 15) * INK;
    const __bf16* brow = W0 + (size_t)n * INK;
#pragma unroll
    for (int kt = 0; kt < INK / 32; ++kt)
      acc = wmma_bf16(load_frag(arow, kt, lane), load_frag(brow, kt, lane), acc);
    float bias = b0[n];
#pragma unroll
    for (int i = 0; i < 8; ++i) {
      float v = acc[i] + bias;
      v = v > 0.f ? v : 0.f;
      sT1[(rbase + i) * DHID + n] = (__bf16)v;
    }
  }
  __syncthreads();

  // layer 2: sT1 * W1^T -> relu -> sT2
  {
    v8f acc = {0.f, 0.f, 0.f, 0.f, 0.f, 0.f, 0.f, 0.f};
    const __bf16* arow = sT1 + (size_t)(lane & 15) * DHID;
    const __bf16* brow = W1 + (size_t)n * DHID;
#pragma unroll
    for (int kt = 0; kt < DHID / 32; ++kt)
      acc = wmma_bf16(load_frag(arow, kt, lane), load_frag(brow, kt, lane), acc);
    float bias = b1[n];
#pragma unroll
    for (int i = 0; i < 8; ++i) {
      float v = acc[i] + bias;
      v = v > 0.f ? v : 0.f;
      sT2[(rbase + i) * DHID + n] = (__bf16)v;
    }
  }
  __syncthreads();

  // layer 3: sT2 * W2^T -> atomic segment-sum scatter to msg[dst]
  {
    v8f acc = {0.f, 0.f, 0.f, 0.f, 0.f, 0.f, 0.f, 0.f};
    const __bf16* arow = sT2 + (size_t)(lane & 15) * DHID;
    const __bf16* brow = W2 + (size_t)n * DHID;
#pragma unroll
    for (int kt = 0; kt < DHID / 32; ++kt)
      acc = wmma_bf16(load_frag(arow, kt, lane), load_frag(brow, kt, lane), acc);
    float bias = b2[n];
#pragma unroll
    for (int i = 0; i < 8; ++i) {
      int m = rbase + i;
      int d = sDst[m];
      if (d >= 0) atomicAdd(&msg[(size_t)d * DHID + n], acc[i] + bias);
    }
  }
}

// ---- GRU over one level's nodes -------------------------------------------
// 16 nodes per block. gi = [msg|x][16x192] * Wih^T (384 cols),
// gh = h[16x128] * Whh^T (384 cols); each wave owns 3 column tiles.
__global__ __launch_bounds__(256) void k_gru(
    int level, const int* c, const int* lvlNodes, const float* msg,
    const float* x, const __bf16* Wih, const __bf16* Whh, const float* bih,
    const float* bhh, float* h_f32, __bf16* h_bf) {
  __shared__ int sNode[16];
  __shared__ __bf16 sXC[16 * 192];
  __shared__ __bf16 sH[16 * DHID];
  __shared__ float sGI[16 * 384];
  __shared__ float sGH[16 * 384];

  const int off = c[C_OFFN + level];
  const int cnt = c[C_OFFN + level + 1] - off;
  const int t = blockIdx.x;
  if (t * 16 >= cnt) return;

  const int tid = threadIdx.x;
  const int wave = tid >> 5;
  const int lane = tid & 31;

  if (tid < 16) {
    int e = t * 16 + tid;
    sNode[tid] = (e < cnt) ? lvlNodes[off + e] : -1;
  }
  __syncthreads();

  // stage xcat = [msg(128) | x(64)] and h (bf16)
#pragma unroll
  for (int p = tid; p < 16 * 192; p += 256) {
    int m = p / 192, k = p % 192;
    int nd = sNode[m];
    float v = 0.f;
    if (nd >= 0)
      v = (k < DHID) ? msg[(size_t)nd * DHID + k]
                     : x[(size_t)nd * DXF + (k - DHID)];
    sXC[p] = (__bf16)v;
  }
#pragma unroll
  for (int p = tid; p < 16 * DHID; p += 256) {
    int m = p / DHID, k = p % DHID;
    int nd = sNode[m];
    sH[p] = (nd >= 0) ? h_bf[(size_t)nd * DHID + k] : (__bf16)0.f;
  }
  __syncthreads();

  const int rbase = (lane < 16) ? 0 : 8;
#pragma unroll
  for (int ci = 0; ci < 3; ++ci) {
    int ct = wave + ci * 8;           // 0..23 column tiles of 384
    int n = ct * 16 + (lane & 15);
    // gi (K = 192)
    {
      v8f acc = {0.f, 0.f, 0.f, 0.f, 0.f, 0.f, 0.f, 0.f};
      const __bf16* arow = sXC + (size_t)(lane & 15) * 192;
      const __bf16* brow = Wih + (size_t)n * 192;
#pragma unroll
      for (int kt = 0; kt < 6; ++kt)
        acc = wmma_bf16(load_frag(arow, kt, lane), load_frag(brow, kt, lane), acc);
      float bias = bih[n];
#pragma unroll
      for (int i = 0; i < 8; ++i) sGI[(rbase + i) * 384 + n] = acc[i] + bias;
    }
    // gh (K = 128)
    {
      v8f acc = {0.f, 0.f, 0.f, 0.f, 0.f, 0.f, 0.f, 0.f};
      const __bf16* arow = sH + (size_t)(lane & 15) * DHID;
      const __bf16* brow = Whh + (size_t)n * DHID;
#pragma unroll
      for (int kt = 0; kt < 4; ++kt)
        acc = wmma_bf16(load_frag(arow, kt, lane), load_frag(brow, kt, lane), acc);
      float bias = bhh[n];
#pragma unroll
      for (int i = 0; i < 8; ++i) sGH[(rbase + i) * 384 + n] = acc[i] + bias;
    }
  }
  __syncthreads();

  // fused gate math, update h in place (f32 output + bf16 mirror)
#pragma unroll
  for (int p = tid; p < 16 * DHID; p += 256) {
    int m = p / DHID, j = p % DHID;
    int nd = sNode[m];
    if (nd < 0) continue;
    float r = sigmoidf_(sGI[m * 384 + j] + sGH[m * 384 + j]);
    float z = sigmoidf_(sGI[m * 384 + DHID + j] + sGH[m * 384 + DHID + j]);
    float nn = tanhf(sGI[m * 384 + 2 * DHID + j] + r * sGH[m * 384 + 2 * DHID + j]);
    size_t idx = (size_t)nd * DHID + j;
    float h = h_f32[idx];
    float hn = (1.f - z) * nn + z * h;
    h_f32[idx] = hn;
    h_bf[idx] = (__bf16)hn;
  }
}

// ---------------------------------------------------------------------------
extern "C" void kernel_launch(void* const* d_in, const int* in_sizes, int n_in,
                              void* d_out, int out_size, void* d_ws,
                              size_t ws_size, hipStream_t stream) {
  (void)in_sizes; (void)n_in; (void)out_size; (void)ws_size;
  const float* x        = (const float*)d_in[0];
  const int*   eidx     = (const int*)d_in[1];
  const int*   fl       = (const int*)d_in[2];
  const float* hs_init  = (const float*)d_in[3];
  const float* Ws0 = (const float*)d_in[4];  const float* bs0 = (const float*)d_in[5];
  const float* Ws1 = (const float*)d_in[6];  const float* bs1 = (const float*)d_in[7];
  const float* Ws2 = (const float*)d_in[8];  const float* bs2 = (const float*)d_in[9];
  const float* Wf0 = (const float*)d_in[10]; const float* bf0 = (const float*)d_in[11];
  const float* Wf1 = (const float*)d_in[12]; const float* bf1 = (const float*)d_in[13];
  const float* Wf2 = (const float*)d_in[14]; const float* bf2 = (const float*)d_in[15];
  const float* Wih_s = (const float*)d_in[16]; const float* Whh_s = (const float*)d_in[17];
  const float* bih_s = (const float*)d_in[18]; const float* bhh_s = (const float*)d_in[19];
  const float* Wih_f = (const float*)d_in[20]; const float* Whh_f = (const float*)d_in[21];
  const float* bih_f = (const float*)d_in[22]; const float* bhh_f = (const float*)d_in[23];

  const int* esrc = eidx;
  const int* edst = eidx + NEDGE;

  float* hs_f32 = (float*)d_out;                        // output lives in d_out
  float* hf_f32 = hs_f32 + (size_t)NNODE * DHID;

  // workspace carve-out
  char* ws = (char*)d_ws;
  size_t off = 0;
  auto take = [&](size_t bytes) {
    size_t p = off;
    off += (bytes + 255) & ~(size_t)255;
    return p;
  };
  __bf16* hs_bf = (__bf16*)(ws + take((size_t)NNODE * DHID * 2));
  __bf16* ns_bf = (__bf16*)(ws + take((size_t)NNODE * 2 * DHID * 2));
  __bf16* hf_bf = (__bf16*)(ws + take((size_t)NNODE * DHID * 2));
  float* msg_s = (float*)(ws + take((size_t)NNODE * DHID * 4));
  float* msg_f = (float*)(ws + take((size_t)NNODE * DHID * 4));
  __bf16* Ws0b = (__bf16*)(ws + take(128 * 128 * 2));
  __bf16* Ws1b = (__bf16*)(ws + take(128 * 128 * 2));
  __bf16* Ws2b = (__bf16*)(ws + take(128 * 128 * 2));
  __bf16* Wf0b = (__bf16*)(ws + take(128 * 256 * 2));
  __bf16* Wf1b = (__bf16*)(ws + take(128 * 128 * 2));
  __bf16* Wf2b = (__bf16*)(ws + take(128 * 128 * 2));
  __bf16* Wihsb = (__bf16*)(ws + take(384 * 192 * 2));
  __bf16* Whhsb = (__bf16*)(ws + take(384 * 128 * 2));
  __bf16* Wihfb = (__bf16*)(ws + take(384 * 192 * 2));
  __bf16* Whhfb = (__bf16*)(ws + take(384 * 128 * 2));
  int* lvlNodes = (int*)(ws + take((size_t)NNODE * 4));
  int* lvlSrc   = (int*)(ws + take((size_t)NEDGE * 4));
  int* lvlDst   = (int*)(ws + take((size_t)NEDGE * 4));
  int* cnt      = (int*)(ws + take(128 * 4));

  const int TPB = 256;
  const int gN = (NNODE + TPB - 1) / TPB;
  const int gE = (NEDGE + TPB - 1) / TPB;
  const int EDGE_TILES = (NEDGE + 15) / 16;
  const int NODE_TILES = (NNODE + 15) / 16;

  // partition graph by destination level
  k_zero_counts<<<1, 128, 0, stream>>>(cnt);
  k_count<<<gE, TPB, 0, stream>>>(fl, esrc, edst, cnt);
  k_offsets<<<1, 1, 0, stream>>>(cnt);
  k_fill<<<gE, TPB, 0, stream>>>(fl, esrc, edst, cnt, lvlNodes, lvlSrc, lvlDst);

  // init state + bf16 mirrors + zero msg buffers
  k_init<<<gN, TPB, 0, stream>>>(hs_init, fl, hs_f32, hf_f32, hs_bf, hf_bf,
                                 ns_bf, msg_s, msg_f);

  // weight conversion f32 -> bf16
  auto cvt = [&](const float* s, __bf16* d, int n) {
    k_cvt<<<(n + TPB - 1) / TPB, TPB, 0, stream>>>(s, d, n);
  };
  cvt(Ws0, Ws0b, 128 * 128); cvt(Ws1, Ws1b, 128 * 128); cvt(Ws2, Ws2b, 128 * 128);
  cvt(Wf0, Wf0b, 128 * 256); cvt(Wf1, Wf1b, 128 * 128); cvt(Wf2, Wf2b, 128 * 128);
  cvt(Wih_s, Wihsb, 384 * 192); cvt(Whh_s, Whhsb, 384 * 128);
  cvt(Wih_f, Wihfb, 384 * 192); cvt(Whh_f, Whhfb, 384 * 128);

  for (int level = 1; level < NLVL; ++level) {
    // structural path: MLP3 over live hs, then GRU update of hs
    k_msg<128><<<EDGE_TILES, TPB, 0, stream>>>(level, cnt, lvlSrc, lvlDst,
                                               hs_bf, Ws0b, bs0, Ws1b, bs1,
                                               Ws2b, bs2, msg_s);
    k_gru<<<NODE_TILES, TPB, 0, stream>>>(level, cnt, lvlNodes, msg_s, x,
                                          Wihsb, Whhsb, bih_s, bhh_s, hs_f32,
                                          hs_bf);
    // functional path: MLP3 over frozen node_state, GRU update of hf
    k_msg<256><<<EDGE_TILES, TPB, 0, stream>>>(level, cnt, lvlSrc, lvlDst,
                                               ns_bf, Wf0b, bf0, Wf1b, bf1,
                                               Wf2b, bf2, msg_f);
    k_gru<<<NODE_TILES, TPB, 0, stream>>>(level, cnt, lvlNodes, msg_f, x,
                                          Wihfb, Whhfb, bih_f, bhh_f, hf_f32,
                                          hf_bf);
  }
}
